// FreeEnergyPredictor_38551626449390
// MI455X (gfx1250) — matrix operand order repaired and checked by reading, not compiled, not run on hardware
//
#include <hip/hip_runtime.h>
#include <hip/hip_bf16.h>
#include <math.h>
#include <stdint.h>

// ---------------- problem constants ----------------
#define HIDDEN   128
#define FILTERS  128
#define NGAUSS   50
#define NITER    6
#define NNODES   16384          // G*A
#define NGRAPH   256
#define APG      64             // atoms per graph
#define EPER     4096           // edges per graph
#define NEDGE    (NGRAPH * EPER)
#define KP1      64             // rbf K padded 50 -> 64
#define ITER_W   (128*KP1 + 4*128*128)   // f16 elems per iteration weight block
// LDS strides (elements), padded so 16B fragment loads are bank-conflict-free
#define SLD      136            // per-wave [16][128] f16 stage
#define HX_LD    136            // [64][128] f16 hx / t1 stage
#define AGG_LD   132            // [64][128] f32 accumulator
#define W1_LD    72             // LDS copy of w1t [128][64] -> stride 72
#define W2_LD    136            // LDS copy of w2t [128][128] -> stride 136
#define LOG2F_   0.6931471805599453f
#define RBF_STEP (10.0f/49.0f)
#define RBF_COEFF (-12.005f)    // -0.5/step^2
#define PI_OVER_CUT 0.314159265358979f

typedef __attribute__((ext_vector_type(16))) _Float16 v16h;
typedef __attribute__((ext_vector_type(8)))  float    v8f;

#define LDS_WAIT()   asm volatile("s_wait_dscnt 0x0" ::: "memory")
#define ASYNC_WAIT() asm volatile("s_wait_asynccnt 0x0" ::: "memory")

// per-lane async 16B copy global -> LDS (tracked by ASYNCcnt)
__device__ __forceinline__ void async_b128(void* lds, const void* g) {
  unsigned lo = (unsigned)(uintptr_t)lds;   // LDS aperture: low 32 bits = LDS offset
  asm volatile("global_load_async_to_lds_b128 %0, %1, off"
               :: "v"(lo), "v"(g) : "memory");
}

__device__ __forceinline__ v8f wmma16(v16h a, v16h b, v8f c) {
  // D = A(16x32 f16) * B(32x16 f16) + C(16x16 f32)
  return __builtin_amdgcn_wmma_f32_16x16x32_f16(false, a, false, b, (short)0, c,
                                                false, false);
}

__device__ __forceinline__ float sspf(float x) {  // shifted softplus
  float sp = (x > 15.0f) ? x : log1pf(expf(x));
  return sp - LOG2F_;
}

// ---- A fragment (16x32 f16): lane = hi*16 + row, element e ->
//      K = kt*32 + (e<8 ? hi*8+e : 16 + hi*8 + (e-8)) -------------------
__device__ __forceinline__ v16h a_frag_g32(const float* rowbase, int ld, int lane, int kt) {
  const int r = lane & 15, hi = lane >> 4;
  const float* p = rowbase + (size_t)r * ld + kt * 32 + hi * 8;
  v16h a;
#pragma unroll
  for (int e = 0; e < 8; ++e) { a[e] = (_Float16)p[e]; a[e + 8] = (_Float16)p[e + 16]; }
  return a;
}
__device__ __forceinline__ v16h a_frag_lds_h(const _Float16* rowbase, int ld, int lane, int kt) {
  const int r = lane & 15, hi = lane >> 4;
  const _Float16* p = rowbase + r * ld + kt * 32 + hi * 8;
  v16h a;
#pragma unroll
  for (int e = 0; e < 8; ++e) { a[e] = p[e]; a[e + 8] = p[e + 16]; }
  return a;
}
__device__ __forceinline__ v16h a_frag_lds_f32(const float* rowbase, int ld, int lane, int kt) {
  const int r = lane & 15, hi = lane >> 4;
  const float* p = rowbase + r * ld + kt * 32 + hi * 8;
  v16h a;
#pragma unroll
  for (int e = 0; e < 8; ++e) { a[e] = (_Float16)p[e]; a[e + 8] = (_Float16)p[e + 16]; }
  return a;
}
// ---- B fragment (32x16 f16) from W^T stored f16 [N][ldk]: lane = kb*16 + col,
//      element e -> K = kt*32 + (e<8 ? kb*8+e : 16 + kb*8 + (e-8)) ----------
__device__ __forceinline__ v16h b_frag(const _Float16* wt, int ldk, int nt, int kt, int lane) {
  const int c = lane & 15, kb = lane >> 4;
  const _Float16* p = wt + (size_t)(nt * 16 + c) * ldk + kt * 32 + kb * 8;
  v16h b;
#pragma unroll
  for (int e = 0; e < 8; ++e) { b[e] = p[e]; b[e + 8] = p[e + 16]; }
  return b;
}

// ---------------- h init: h = emb[z] ----------------
__global__ void init_h_kernel(const int* __restrict__ z, const float* __restrict__ emb,
                              float* __restrict__ h) {
  int i = blockIdx.x * blockDim.x + threadIdx.x;   // N*128 threads
  int node = i >> 7, f = i & 127;
  h[i] = emb[z[node] * 128 + f];
}

// ---------------- weight prep: transpose + f32->f16, pad rbf K to 64 --------
__global__ void prep_w_kernel(const float* __restrict__ w1, const float* __restrict__ w2,
                              const float* __restrict__ cw1, const float* __restrict__ cw2,
                              const float* __restrict__ bw, _Float16* __restrict__ out) {
  int i = blockIdx.x * blockDim.x + threadIdx.x;
  if (i >= NITER * ITER_W) return;
  int it = i / ITER_W, o = i % ITER_W;
  float val;
  if (o < 128 * KP1) {                       // w1t [128][64] from [50][128]
    int n = o >> 6, k = o & 63;
    val = (k < NGAUSS) ? w1[it * NGAUSS * 128 + k * 128 + n] : 0.0f;
  } else {
    o -= 128 * KP1;
    int mtx = o >> 14, oo = o & 16383;       // 4 matrices of [128][128]
    int n = oo >> 7, k = oo & 127;
    const float* src = (mtx == 0) ? w2 : (mtx == 1) ? cw1 : (mtx == 2) ? cw2 : bw;
    val = src[it * 16384 + k * 128 + n];
  }
  out[i] = (_Float16)val;
}

// ---------------- one SchNet interaction, one workgroup per molecule --------
__global__ void __launch_bounds__(256)
schnet_iter_kernel(const float* __restrict__ pos,
                   const int* __restrict__ esrc, const int* __restrict__ edst,
                   float* __restrict__ h,
                   const _Float16* __restrict__ w1t,  const float* __restrict__ b1,
                   const _Float16* __restrict__ w2t,  const float* __restrict__ b2,
                   const _Float16* __restrict__ cw1t,
                   const _Float16* __restrict__ cw2t, const float* __restrict__ cb2,
                   const _Float16* __restrict__ bwt,  const float* __restrict__ bb) {
  __shared__ _Float16 hxLDS[APG * HX_LD];       // phase A: hx ; phase C: t1 stage
  __shared__ float    aggLDS[APG * AGG_LD];     // segment-sum accumulator
  __shared__ _Float16 stage[8 * 16 * SLD];      // per-wave hid1 stage
  __shared__ _Float16 lw1[128 * W1_LD];         // LDS copy of w1t (hot loop)
  __shared__ _Float16 lw2[128 * W2_LD];         // LDS copy of w2t (hot loop)
  __shared__ float    dSh[8 * 16];              // per-wave edge distances
  __shared__ int      sSh[8 * 16];              // per-wave local src
  __shared__ int      tSh[8 * 16];              // per-wave local dst

  const int g    = blockIdx.x;
  const int n0   = g * APG;
  const int e0   = g * EPER;
  const int tid  = threadIdx.x;
  const int w    = tid >> 5;                    // wave id 0..7
  const int lane = tid & 31;
  const int cc   = lane & 15, hi = lane >> 4;

  // ---- async-stage hot-loop weights into LDS (16B per lane per op) ---------
  {
#pragma unroll
    for (int idx = tid; idx < (128 * KP1) / 8; idx += 256) {   // 1024 chunks
      int n = idx >> 3, kc = (idx & 7) * 8;
      async_b128(&lw1[n * W1_LD + kc], w1t + n * KP1 + kc);
    }
#pragma unroll
    for (int idx = tid; idx < (128 * 128) / 8; idx += 256) {   // 2048 chunks
      int n = idx >> 4, kc = (idx & 15) * 8;
      async_b128(&lw2[n * W2_LD + kc], w2t + n * 128 + kc);
    }
  }

  // ---------------- Phase A: hx = h @ conv_w1 (f16 into LDS); zero agg ------
  {
    const int tile = w & 3, nh = w >> 2;        // 4 node tiles x 2 n-halves
    const float* rowbase = h + (size_t)(n0 + tile * 16) * 128;
    v16h ah[4];
#pragma unroll
    for (int kt = 0; kt < 4; ++kt) ah[kt] = a_frag_g32(rowbase, 128, lane, kt);
    for (int nt = nh * 4; nt < nh * 4 + 4; ++nt) {
      v8f acc = {};
#pragma unroll
      for (int kt = 0; kt < 4; ++kt) acc = wmma16(ah[kt], b_frag(cw1t, 128, nt, kt, lane), acc);
#pragma unroll
      for (int v = 0; v < 8; ++v)
        hxLDS[(tile * 16 + v + 8 * hi) * HX_LD + nt * 16 + cc] = (_Float16)acc[v];
    }
    for (int idx = tid; idx < APG * AGG_LD; idx += 256) aggLDS[idx] = 0.0f;
  }
  ASYNC_WAIT();          // weight copies landed in LDS
  __syncthreads();

  // ---------------- Phase B: edge filter network + gather/scatter -----------
  {
    _Float16* st = stage + w * 16 * SLD;
    float* dT = dSh + w * 16; int* sT = sSh + w * 16; int* tT = tSh + w * 16;
    for (int t = w; t < EPER / 16; t += 8) {    // 32 edge tiles per wave
      const int eb = e0 + t * 16;
      // per-lane geometry for row r = lane&15 (both halves redundant)
      const int r  = cc;
      const int gs = esrc[eb + r], gd = edst[eb + r];
      float dx = pos[gs * 3 + 0] - pos[gd * 3 + 0];
      float dy = pos[gs * 3 + 1] - pos[gd * 3 + 1];
      float dz = pos[gs * 3 + 2] - pos[gd * 3 + 2];
      float d  = sqrtf(dx * dx + dy * dy + dz * dz + 1e-12f);
      if (hi == 0) { dT[r] = d; sT[r] = gs - n0; } else { tT[r] = gd - n0; }

      // rbf A fragments (K padded to 64)
      v16h ar[2];
#pragma unroll
      for (int kt = 0; kt < 2; ++kt) {
        v16h a;
#pragma unroll
        for (int e = 0; e < 16; ++e) {
          int k = kt * 32 + ((e < 8) ? hi * 8 + e : 16 + hi * 8 + (e - 8));
          float rv = 0.0f;
          if (k < NGAUSS) { float u = d - (float)k * RBF_STEP; rv = expf(RBF_COEFF * u * u); }
          a[e] = (_Float16)rv;
        }
        ar[kt] = a;
      }
      // mm1: hid1 = ssp(rbf @ w1 + b1) -> f16 stage (B frags from LDS)
      for (int nt = 0; nt < 8; ++nt) {
        v8f acc = {};
        acc = wmma16(ar[0], b_frag(lw1, W1_LD, nt, 0, lane), acc);
        acc = wmma16(ar[1], b_frag(lw1, W1_LD, nt, 1, lane), acc);
        const int n = nt * 16 + cc;
        const float bia = b1[n];
#pragma unroll
        for (int v = 0; v < 8; ++v)
          st[(v + 8 * hi) * SLD + n] = (_Float16)sspf(acc[v] + bia);
      }
      LDS_WAIT();  // stage + dT/sT/tT stores visible (LDS is in-order per wave)

      v16h af[4];
#pragma unroll
      for (int kt = 0; kt < 4; ++kt) af[kt] = a_frag_lds_h(st, SLD, lane, kt);

      float Cm[8]; int sl[8], dl[8];
#pragma unroll
      for (int v = 0; v < 8; ++v) {
        const int m = v + 8 * hi;
        Cm[v] = 0.5f * (cosf(dT[m] * PI_OVER_CUT) + 1.0f);
        sl[v] = sT[m]; dl[v] = tT[m];
      }
      // mm2: Wf = (hid1 @ w2 + b2) * C ; then agg[dst] += hx[src] * Wf
      for (int nt = 0; nt < 8; ++nt) {
        v8f acc = {};
#pragma unroll
        for (int kt = 0; kt < 4; ++kt)
          acc = wmma16(af[kt], b_frag(lw2, W2_LD, nt, kt, lane), acc);
        const int n = nt * 16 + cc;
        const float bia = b2[n];
#pragma unroll
        for (int v = 0; v < 8; ++v) {
          float wf = (acc[v] + bia) * Cm[v];
          float xj = (float)hxLDS[sl[v] * HX_LD + n];
          atomicAdd(&aggLDS[dl[v] * AGG_LD + n], wf * xj);   // ds_add_f32
        }
      }
    }
  }
  __syncthreads();

  // ---------------- Phase C: node update, h += ssp(agg@W2+b)@blk_w + bb -----
  {
    const int tile = w & 3, nh = w >> 2;
    v16h aa[4];
#pragma unroll
    for (int kt = 0; kt < 4; ++kt)
      aa[kt] = a_frag_lds_f32(aggLDS + tile * 16 * AGG_LD, AGG_LD, lane, kt);
    for (int nt = nh * 4; nt < nh * 4 + 4; ++nt) {
      v8f acc = {};
#pragma unroll
      for (int kt = 0; kt < 4; ++kt) acc = wmma16(aa[kt], b_frag(cw2t, 128, nt, kt, lane), acc);
      const int n = nt * 16 + cc;
      const float bia = cb2[n];
#pragma unroll
      for (int v = 0; v < 8; ++v)
        hxLDS[(tile * 16 + v + 8 * hi) * HX_LD + n] = (_Float16)sspf(acc[v] + bia);
    }
    __syncthreads();
    v16h at[4];
#pragma unroll
    for (int kt = 0; kt < 4; ++kt)
      at[kt] = a_frag_lds_h(hxLDS + tile * 16 * HX_LD, HX_LD, lane, kt);
    for (int nt = nh * 4; nt < nh * 4 + 4; ++nt) {
      v8f acc = {};
#pragma unroll
      for (int kt = 0; kt < 4; ++kt) acc = wmma16(at[kt], b_frag(bwt, 128, nt, kt, lane), acc);
      const int n = nt * 16 + cc;
      const float bia = bb[n];
#pragma unroll
      for (int v = 0; v < 8; ++v) {
        size_t idx = (size_t)(n0 + tile * 16 + v + 8 * hi) * 128 + n;
        h[idx] = h[idx] + acc[v] + bia;        // residual (rows block-private)
      }
    }
  }
}

// ---------------- mean pool + linear head ----------------
__global__ void pool_head_kernel(const float* __restrict__ h, const float* __restrict__ hw,
                                 const float* __restrict__ hb, float* __restrict__ out) {
  __shared__ float red[128];
  const int g = blockIdx.x, f = threadIdx.x;   // 128 threads
  const float* base = h + (size_t)g * APG * 128;
  float s = 0.0f;
  for (int j = 0; j < APG; ++j) s += base[j * 128 + f];
  red[f] = (s * (1.0f / (float)APG)) * hw[f];
  __syncthreads();
  for (int off = 64; off > 0; off >>= 1) {
    if (f < off) red[f] += red[f + off];
    __syncthreads();
  }
  if (f == 0) out[g] = red[0] + hb[0];
}

extern "C" void kernel_launch(void* const* d_in, const int* in_sizes, int n_in,
                              void* d_out, int out_size, void* d_ws, size_t ws_size,
                              hipStream_t stream) {
  const int*   z       = (const int*)d_in[0];
  const float* pos     = (const float*)d_in[1];
  // d_in[2] batch: implicit (graphs are contiguous blocks of 64)
  const int*   eidx    = (const int*)d_in[3];     // [2,E]
  const float* emb     = (const float*)d_in[4];
  const float* mlp_w1  = (const float*)d_in[5];
  const float* mlp_b1  = (const float*)d_in[6];
  const float* mlp_w2  = (const float*)d_in[7];
  const float* mlp_b2  = (const float*)d_in[8];
  const float* conv_w1 = (const float*)d_in[9];
  const float* conv_w2 = (const float*)d_in[10];
  const float* conv_b2 = (const float*)d_in[11];
  const float* blk_w   = (const float*)d_in[12];
  const float* blk_b   = (const float*)d_in[13];
  const float* head_w  = (const float*)d_in[14];
  const float* head_b  = (const float*)d_in[15];

  float*     h    = (float*)d_ws;                                   // [N,128] f32
  _Float16*  wbuf = (_Float16*)((char*)d_ws + (size_t)NNODES * 128 * 4);

  init_h_kernel<<<(NNODES * 128) / 256, 256, 0, stream>>>(z, emb, h);
  prep_w_kernel<<<(NITER * ITER_W + 255) / 256, 256, 0, stream>>>(
      mlp_w1, mlp_w2, conv_w1, conv_w2, blk_w, wbuf);

  const int*   esrc = eidx;
  const int*   edst = eidx + NEDGE;
  for (int i = 0; i < NITER; ++i) {
    const _Float16* wb = wbuf + (size_t)i * ITER_W;
    schnet_iter_kernel<<<NGRAPH, 256, 0, stream>>>(
        pos, esrc, edst, h,
        /*w1t*/ wb,                      mlp_b1 + i * 128,
        /*w2t*/ wb + 128 * KP1,          mlp_b2 + i * 128,
        /*cw1t*/ wb + 128 * KP1 + 16384,
        /*cw2t*/ wb + 128 * KP1 + 32768, conv_b2 + i * 128,
        /*bwt*/  wb + 128 * KP1 + 49152, blk_b + i * 128);
  }
  pool_head_kernel<<<NGRAPH, 128, 0, stream>>>(h, head_w, head_b, (float*)d_out);
}
